// KimiDeltaAttention_18365280157742
// MI455X (gfx1250) — compile-verified, matrix operand order checked by reading, not run on hardware
//
#include <hip/hip_runtime.h>

// ---------------------------------------------------------------------------
// Kimi delta-attention block for MI455X (gfx1250, wave32, WMMA).
// Big GEMMs: v_wmma_f32_16x16x32_bf16, double-buffered LDS, async global->LDS
// copies (ASYNCcnt) when available. Recurrent scan keeps the 128x128 per-head
// state resident in LDS (320KB/WGP on CDNA5).
// ---------------------------------------------------------------------------

typedef unsigned short u16;
typedef __attribute__((ext_vector_type(16))) __bf16 v16bf;
typedef __attribute__((ext_vector_type(8)))  float  v8f;
typedef int kda_v4i __attribute__((vector_size(16)));   // matches async builtin

#define KDA_N     2048
#define KDA_HID   2048
#define KDA_H     32
#define KDA_DK    128
#define KDA_PROJ  4096   // H*DK
#define KDA_CONV  4

#define KDA_AS1 __attribute__((address_space(1)))
#define KDA_AS3 __attribute__((address_space(3)))

#if __has_builtin(__builtin_amdgcn_global_load_async_to_lds_b128)
#define KDA_USE_ASYNC 1
#else
#define KDA_USE_ASYNC 0
#endif

// ------------------------------ helpers ------------------------------------

__device__ __forceinline__ u16 f32_to_bf16_rne(float f) {
  unsigned int u = __float_as_uint(f);
  u = (u + 0x7FFFu + ((u >> 16) & 1u)) >> 16;
  return (u16)u;
}

__device__ __forceinline__ float sigmoidf_dev(float x) {
  return 1.0f / (1.0f + __expf(-x));
}

__device__ __forceinline__ void kda_wait_async8() {
#if KDA_USE_ASYNC
#if __has_builtin(__builtin_amdgcn_s_wait_asynccnt)
  __builtin_amdgcn_s_wait_asynccnt(8);
#else
  asm volatile("s_wait_asynccnt 0x8" ::: "memory");
#endif
#endif
}

__device__ __forceinline__ void kda_wait_async0() {
#if KDA_USE_ASYNC
#if __has_builtin(__builtin_amdgcn_s_wait_asynccnt)
  __builtin_amdgcn_s_wait_asynccnt(0);
#else
  asm volatile("s_wait_asynccnt 0x0" ::: "memory");
#endif
#endif
}

// f32 -> bf16 (stored as ushort) conversion
__global__ void kda_cvt_bf16(const float* __restrict__ in, u16* __restrict__ out, int n) {
  int i = blockIdx.x * blockDim.x + threadIdx.x;
  if (i < n) out[i] = f32_to_bf16_rne(in[i]);
}

// --------------------------- WMMA NT GEMM ----------------------------------
// C[M,N] = A[M,K] * B[N,K]^T ; A,B bf16 (ushort), C f32.  K multiple of 32.
// Block 128 threads = 4 waves; BM=BN=128, BK=32; wave -> 64x64 macro-tile
// (16 wmma per 8 fragment loads). Double-buffered LDS, pipelined loads.

#define GBM 128
#define GBN 128
#define GBK 32
#define GLD 40   // LDS row stride in ushorts (32 + 8 pad); 80B rows, 16B aligned

union KdaFrag { v16bf v; uint4 q[2]; };

__global__ __launch_bounds__(128) void kda_gemm_nt_bf16(
    const u16* __restrict__ A, const u16* __restrict__ B, float* __restrict__ C,
    int M, int N, int K)
{
  __shared__ __align__(16) u16 lA[2][GBM * GLD];
  __shared__ __align__(16) u16 lB[2][GBN * GLD];

  const int tid  = threadIdx.x;      // 0..127
  const int lane = tid & 31;
  const int wave = tid >> 5;         // 0..3
  const int wr   = wave >> 1;        // 0..1 : M (64-row half)
  const int wc   = wave & 1;         // 0..1 : N (64-col half)
  const int blockM = blockIdx.y * GBM;
  const int blockN = blockIdx.x * GBN;

  // Clamp source rows instead of predicating loads: OOB tiles contribute
  // garbage to accumulators that the guarded epilogue never writes. This
  // keeps the hot loop free of exec-mask manipulation.
  const int rowA = min(blockM + tid, M - 1);
  const int rowB = min(blockN + tid, N - 1);
  const u16* __restrict__ pA = A + (size_t)rowA * K;
  const u16* __restrict__ pB = B + (size_t)rowB * K;

  v8f acc[4][4];
  {
    v8f zero = {0.f, 0.f, 0.f, 0.f, 0.f, 0.f, 0.f, 0.f};
#pragma unroll
    for (int i = 0; i < 4; ++i)
#pragma unroll
      for (int j = 0; j < 4; ++j) acc[i][j] = zero;
  }

  const int fragRow = lane & 15;
  const int kSel    = (lane >> 4) * 8;    // 0 or 8 (ushorts)
  const int T       = K / GBK;

  // each thread stages one 32-ushort (64B) row of each tile = 4x b128 each
#if !KDA_USE_ASYNC
  uint4 ra[4], rb[4];
#endif

  auto compute = [&](int buf) {
    KdaFrag afr[4], bfr[4];
#pragma unroll
    for (int mt = 0; mt < 4; ++mt) {
      const u16* p = &lA[buf][(wr * 64 + mt * 16 + fragRow) * GLD + kSel];
      afr[mt].q[0] = *(const uint4*)p;
      afr[mt].q[1] = *(const uint4*)(p + 16);
    }
#pragma unroll
    for (int nt = 0; nt < 4; ++nt) {
      const u16* p = &lB[buf][(wc * 64 + nt * 16 + fragRow) * GLD + kSel];
      bfr[nt].q[0] = *(const uint4*)p;
      bfr[nt].q[1] = *(const uint4*)(p + 16);
    }
#pragma unroll
    for (int mt = 0; mt < 4; ++mt)
#pragma unroll
      for (int nt = 0; nt < 4; ++nt)
        acc[mt][nt] = __builtin_amdgcn_wmma_f32_16x16x32_bf16(
            false, afr[mt].v, false, bfr[nt].v,
            (short)0, acc[mt][nt], false, false);
  };

#if KDA_USE_ASYNC
  // ---- CDNA5 async global->LDS copies, tracked by ASYNCcnt ----
  auto issue_async = [&](int k0, int buf) {
#pragma unroll
    for (int j = 0; j < 4; ++j) {
      __builtin_amdgcn_global_load_async_to_lds_b128(
          (KDA_AS1 kda_v4i*)(pA + k0 + j * 8),
          (KDA_AS3 kda_v4i*)&lA[buf][tid * GLD + j * 8], 0, 0);
      __builtin_amdgcn_global_load_async_to_lds_b128(
          (KDA_AS1 kda_v4i*)(pB + k0 + j * 8),
          (KDA_AS3 kda_v4i*)&lB[buf][tid * GLD + j * 8], 0, 0);
    }
  };

  issue_async(0, 0);
  for (int i = 0; i < T; ++i) {
    if (i + 1 < T) {
      issue_async((i + 1) * GBK, (i + 1) & 1);  // overlap with tile i compute
      kda_wait_async8();                        // tile i's 8 ops done (in-order)
    } else {
      kda_wait_async0();
    }
    __syncthreads();       // all waves' async writes visible
    compute(i & 1);
    __syncthreads();       // buffer free for reuse at iteration i+2
  }
#else
  // ---- fallback: register-staged double buffering, 1 barrier/iteration ----
  auto gload = [&](int k0) {
#pragma unroll
    for (int j = 0; j < 4; ++j) {
      ra[j] = *(const uint4*)(pA + k0 + j * 8);
      rb[j] = *(const uint4*)(pB + k0 + j * 8);
    }
  };
  auto lstore = [&](int buf) {
#pragma unroll
    for (int j = 0; j < 4; ++j) {
      *(uint4*)(&lA[buf][tid * GLD + j * 8]) = ra[j];
      *(uint4*)(&lB[buf][tid * GLD + j * 8]) = rb[j];
    }
  };

  gload(0);
  lstore(0);
  __syncthreads();
  for (int i = 0; i < T; ++i) {
    if (i + 1 < T) gload((i + 1) * GBK);   // global latency hidden by wmma
    compute(i & 1);
    if (i + 1 < T) lstore((i + 1) & 1);    // other buffer: free since barrier i-1
    __syncthreads();
  }
#endif

  // ---- epilogue: C/D layout -> lane(l): N = l%16, M = vgpr + 8*(l/16) ----
  const int colInTile = lane & 15;
  const int rowHalf   = (lane >> 4) * 8;
#pragma unroll
  for (int mt = 0; mt < 4; ++mt)
#pragma unroll
    for (int nt = 0; nt < 4; ++nt) {
      int col = blockN + wc * 64 + nt * 16 + colInTile;
      if (col >= N) continue;
      int rbase = blockM + wr * 64 + mt * 16 + rowHalf;
#pragma unroll
      for (int r = 0; r < 8; ++r) {
        int row = rbase + r;
        if (row < M) C[(size_t)row * N + col] = acc[mt][nt][r];
      }
    }
}

// ----------------------- depthwise causal conv + SiLU ----------------------
// y[t,d] = silu( sum_{j=0..3} x[t+j-3, d] * w[d, j] )
__global__ void kda_conv_silu(const float* __restrict__ x, const float* __restrict__ w,
                              float* __restrict__ y, int n, int D) {
  int i = blockIdx.x * blockDim.x + threadIdx.x;
  if (i >= n * D) return;
  int t = i / D, d = i - t * D;
  float a = 0.f;
#pragma unroll
  for (int j = 0; j < KDA_CONV; ++j) {
    int tt = t + j - (KDA_CONV - 1);
    float xv = (tt >= 0) ? x[(size_t)tt * D + d] : 0.f;
    a = fmaf(xv, w[d * KDA_CONV + j], a);
  }
  y[i] = a * sigmoidf_dev(a);
}

// ------------- decay gate: g = -exp(A_log[h]) * softplus(f + dt_bias) ------
__global__ void kda_gate_g(float* __restrict__ f, const float* __restrict__ dt_bias,
                           const float* __restrict__ A_log, int total) {
  int i = blockIdx.x * blockDim.x + threadIdx.x;
  if (i >= total) return;
  int j = i & (KDA_PROJ - 1);          // PROJ = 4096 (pow2)
  int h = j >> 7;                      // / DK
  float x = f[i] + dt_bias[j];
  float sp = (x > 20.f) ? x : log1pf(__expf(x));
  f[i] = -__expf(A_log[h]) * sp;
}

__global__ void kda_sigmoid_inplace(float* __restrict__ x, int n) {
  int i = blockIdx.x * blockDim.x + threadIdx.x;
  if (i < n) x[i] = sigmoidf_dev(x[i]);
}

// ---------------- per-(token,head) L2 normalize over DK=128 ----------------
__global__ __launch_bounds__(128) void kda_l2norm(float* __restrict__ x, float scale) {
  __shared__ float red[4];
  int row = blockIdx.x;             // n*H rows
  int d   = threadIdx.x;            // 0..127
  float v = x[(size_t)row * KDA_DK + d];
  float s = v * v;
#pragma unroll
  for (int off = 16; off > 0; off >>= 1) s += __shfl_down(s, off, 32);
  if ((d & 31) == 0) red[d >> 5] = s;
  __syncthreads();
  float tot = red[0] + red[1] + red[2] + red[3];
  x[(size_t)row * KDA_DK + d] = v * rsqrtf(tot + 1e-6f) * scale;
}

// --------------------------- delta-rule scan -------------------------------
// One block per head; 128 threads; thread owns one v-column of S[128][128]
// kept in LDS (64KB + vectors; CDNA5 WGP has 320KB LDS).
__global__ __launch_bounds__(128) void kda_scan(
    const float* __restrict__ q, const float* __restrict__ k,
    const float* __restrict__ v, const float* __restrict__ g,
    const float* __restrict__ beta, float* __restrict__ core, int n)
{
  extern __shared__ float smem[];
  float* S   = smem;                     // 128*128
  float* ks  = S + KDA_DK * KDA_DK;      // 128
  float* qs  = ks + KDA_DK;              // 128
  float* dec = qs + KDA_DK;              // 128

  const int h   = blockIdx.x;
  const int tid = threadIdx.x;           // v-column

  for (int i = tid; i < KDA_DK * KDA_DK; i += KDA_DK) S[i] = 0.f;
  __syncthreads();

  for (int t = 0; t < n; ++t) {
    size_t base = ((size_t)t * KDA_H + h) * KDA_DK;
    float qv = q[base + tid];
    float kv = k[base + tid];
    float vv = v[base + tid];
    float gv = g[base + tid];
    ks[tid]  = kv;
    qs[tid]  = qv;
    dec[tid] = __expf(gv);
    float bt = beta[(size_t)t * KDA_H + h];
    __syncthreads();

    // pass 1: decay state column + memory readout  kv = k^T S
    float kvacc = 0.f;
#pragma unroll 4
    for (int kk = 0; kk < KDA_DK; ++kk) {
      float s = S[kk * KDA_DK + tid] * dec[kk];
      S[kk * KDA_DK + tid] = s;
      kvacc = fmaf(ks[kk], s, kvacc);
    }
    float delta = (vv - kvacc) * bt;

    // pass 2: rank-1 update + output  o = q^T S
    float o = 0.f;
#pragma unroll 4
    for (int kk = 0; kk < KDA_DK; ++kk) {
      float s = S[kk * KDA_DK + tid] + ks[kk] * delta;
      S[kk * KDA_DK + tid] = s;
      o = fmaf(qs[kk], s, o);
    }
    core[base + tid] = o;
    __syncthreads();                    // protect ks/qs/dec for next token
  }
}

// --------- gated RMSNorm (sigmoid gate) -> bf16 for final projection -------
__global__ __launch_bounds__(128) void kda_gated_rms(
    const float* __restrict__ core, const float* __restrict__ gate,
    const float* __restrict__ w, u16* __restrict__ out)
{
  __shared__ float red[4];
  int row = blockIdx.x;              // n*H rows
  int d   = threadIdx.x;
  size_t idx = (size_t)row * KDA_DK + d;
  float c = core[idx];
  float s = c * c;
#pragma unroll
  for (int off = 16; off > 0; off >>= 1) s += __shfl_down(s, off, 32);
  if ((d & 31) == 0) red[d >> 5] = s;
  __syncthreads();
  float ms = (red[0] + red[1] + red[2] + red[3]) * (1.0f / KDA_DK);
  float r  = c * rsqrtf(ms + 1e-5f) * w[d] * sigmoidf_dev(gate[idx]);
  out[idx] = f32_to_bf16_rne(r);
}

// ------------------------------ launcher -----------------------------------

static inline void launch_gemm(const u16* A, const u16* B, float* C,
                               int M, int N, int K, hipStream_t stream) {
  dim3 grid((N + GBN - 1) / GBN, (M + GBM - 1) / GBM);
  kda_gemm_nt_bf16<<<grid, 128, 0, stream>>>(A, B, C, M, N, K);
}

static inline void launch_cvt(const float* in, u16* out, int n, hipStream_t stream) {
  kda_cvt_bf16<<<(n + 255) / 256, 256, 0, stream>>>(in, out, n);
}

extern "C" void kernel_launch(void* const* d_in, const int* in_sizes, int n_in,
                              void* d_out, int out_size, void* d_ws, size_t ws_size,
                              hipStream_t stream) {
  (void)in_sizes; (void)n_in; (void)out_size; (void)ws_size;

  const float* hs      = (const float*)d_in[0];
  const float* Wq      = (const float*)d_in[1];
  const float* Wk      = (const float*)d_in[2];
  const float* Wv      = (const float*)d_in[3];
  const float* conv_q  = (const float*)d_in[4];
  const float* conv_k  = (const float*)d_in[5];
  const float* conv_v  = (const float*)d_in[6];
  const float* Wfa     = (const float*)d_in[7];
  const float* Wfb     = (const float*)d_in[8];
  const float* dt_bias = (const float*)d_in[9];
  const float* Wb      = (const float*)d_in[10];
  const float* A_log   = (const float*)d_in[11];
  const float* Wga     = (const float*)d_in[12];
  const float* Wgb     = (const float*)d_in[13];
  const float* onorm_w = (const float*)d_in[14];
  const float* Wo      = (const float*)d_in[15];
  float* out = (float*)d_out;

  const int N = KDA_N, HID = KDA_HID, H = KDA_H, DK = KDA_DK, PROJ = KDA_PROJ;

  // ---- workspace carve-out (~290 MB with reuse) ----
  char* wsp = (char*)d_ws;
  auto carve = [&](size_t bytes) -> void* {
    void* p = (void*)wsp;
    wsp += (bytes + 255) & ~(size_t)255;
    return p;
  };
  u16* hs_bf  = (u16*)carve((size_t)N * HID * 2);
  u16* wq_bf  = (u16*)carve((size_t)PROJ * HID * 2);
  u16* wk_bf  = (u16*)carve((size_t)PROJ * HID * 2);
  u16* wv_bf  = (u16*)carve((size_t)PROJ * HID * 2);
  u16* wo_bf  = (u16*)carve((size_t)HID * PROJ * 2);
  u16* wfa_bf = (u16*)carve((size_t)DK * HID * 2);
  u16* wfb_bf = (u16*)carve((size_t)PROJ * DK * 2);
  u16* wb_bf  = (u16*)carve((size_t)H * HID * 2);
  u16* wga_bf = (u16*)carve((size_t)DK * HID * 2);
  u16* wgb_bf = (u16*)carve((size_t)PROJ * DK * 2);
  float* qpre = (float*)carve((size_t)N * PROJ * 4);   // later reused as f/g
  float* kpre = (float*)carve((size_t)N * PROJ * 4);   // later reused as gate
  float* vpre = (float*)carve((size_t)N * PROJ * 4);   // later reused as core
  float* qc   = (float*)carve((size_t)N * PROJ * 4);
  float* kc   = (float*)carve((size_t)N * PROJ * 4);
  float* vc   = (float*)carve((size_t)N * PROJ * 4);
  float* lr_t = (float*)carve((size_t)N * DK * 4);     // fa_tmp / ga_tmp
  u16*   lr_b = (u16*)carve((size_t)N * DK * 2);       // bf16 of low-rank tmp
  float* beta = (float*)carve((size_t)N * H * 4);
  u16*   pout = (u16*)carve((size_t)N * PROJ * 2);

  // ---- 1. bf16 conversions ----
  launch_cvt(hs,  hs_bf,  N * HID,    stream);
  launch_cvt(Wq,  wq_bf,  PROJ * HID, stream);
  launch_cvt(Wk,  wk_bf,  PROJ * HID, stream);
  launch_cvt(Wv,  wv_bf,  PROJ * HID, stream);
  launch_cvt(Wo,  wo_bf,  HID * PROJ, stream);
  launch_cvt(Wfa, wfa_bf, DK * HID,   stream);
  launch_cvt(Wfb, wfb_bf, PROJ * DK,  stream);
  launch_cvt(Wb,  wb_bf,  H * HID,    stream);
  launch_cvt(Wga, wga_bf, DK * HID,   stream);
  launch_cvt(Wgb, wgb_bf, PROJ * DK,  stream);

  // ---- 2. Q/K/V projections (WMMA) ----
  launch_gemm(hs_bf, wq_bf, qpre, N, PROJ, HID, stream);
  launch_gemm(hs_bf, wk_bf, kpre, N, PROJ, HID, stream);
  launch_gemm(hs_bf, wv_bf, vpre, N, PROJ, HID, stream);

  // ---- 3. causal depthwise conv + SiLU ----
  {
    int total = N * PROJ, blocks = (total + 255) / 256;
    kda_conv_silu<<<blocks, 256, 0, stream>>>(qpre, conv_q, qc, N, PROJ);
    kda_conv_silu<<<blocks, 256, 0, stream>>>(kpre, conv_k, kc, N, PROJ);
    kda_conv_silu<<<blocks, 256, 0, stream>>>(vpre, conv_v, vc, N, PROJ);
  }

  // ---- 4. low-rank decay path: f = (hs@Wfa^T)@Wfb^T ; g elementwise ----
  launch_gemm(hs_bf, wfa_bf, lr_t, N, DK, HID, stream);
  launch_cvt(lr_t, lr_b, N * DK, stream);
  launch_gemm(lr_b, wfb_bf, qpre /* f, qpre now dead */, N, PROJ, DK, stream);
  kda_gate_g<<<(N * PROJ + 255) / 256, 256, 0, stream>>>(qpre, dt_bias, A_log, N * PROJ);

  // ---- 5. beta = sigmoid(hs @ Wb^T) ----
  launch_gemm(hs_bf, wb_bf, beta, N, H, HID, stream);
  kda_sigmoid_inplace<<<(N * H + 255) / 256, 256, 0, stream>>>(beta, N * H);

  // ---- 6. L2 norm (q scaled by dk^-0.5) ----
  kda_l2norm<<<N * H, 128, 0, stream>>>(qc, rsqrtf((float)DK));
  kda_l2norm<<<N * H, 128, 0, stream>>>(kc, 1.0f);

  // ---- 7. low-rank gate path (independent of scan) ----
  launch_gemm(hs_bf, wga_bf, lr_t, N, DK, HID, stream);
  launch_cvt(lr_t, lr_b, N * DK, stream);
  launch_gemm(lr_b, wgb_bf, kpre /* gate, kpre now dead */, N, PROJ, DK, stream);

  // ---- 8. sequential delta-rule scan (state in LDS) ----
  {
    size_t smem = (size_t)(KDA_DK * KDA_DK + 3 * KDA_DK) * sizeof(float); // 67072 B
    kda_scan<<<H, KDA_DK, smem, stream>>>(qc, kc, vc, qpre /*g*/, beta,
                                          vpre /* core, vpre now dead */, N);
  }

  // ---- 9. gated RMSNorm -> bf16 ----
  kda_gated_rms<<<N * H, 128, 0, stream>>>(vpre /*core*/, kpre /*gate*/, onorm_w, pout);

  // ---- 10. output projection: out = pre @ Wo^T (WMMA) ----
  launch_gemm(pout, wo_bf, out, N, HID, PROJ, stream);
}
